// GNN_Node_9792525435067
// MI455X (gfx1250) — matrix operand order repaired and checked by reading, not compiled, not run on hardware
//
#include <hip/hip_runtime.h>
#include <hip/hip_bf16.h>

typedef __attribute__((ext_vector_type(16))) __bf16 v16bf;
typedef __attribute__((ext_vector_type(8)))  __bf16 v8bf;
typedef __attribute__((ext_vector_type(8)))  float  v8f;

#define NN 200000
#define EE 500000
#define DD 128
#define LL 5
#define LDW (DD + 8)   // padded LDS row stride: 272B = 68 dwords = 4 mod 64 banks

// ---------------- atom encoder layer 1: [N,12] @ [12,128] + BN + ReLU -> bf16
__global__ void atom1_kernel(const float* __restrict__ x, const float* __restrict__ W,
                             const float* __restrict__ b, const float* __restrict__ g,
                             const float* __restrict__ be, const float* __restrict__ m,
                             const float* __restrict__ v, __bf16* __restrict__ outb, int nrows) {
    int idx = blockIdx.x * 256 + threadIdx.x;
    if (idx >= nrows * DD) return;
    int n = idx >> 7, d = idx & 127;
    float acc = b[d];
    const float* xr = x + n * 12;
#pragma unroll
    for (int k = 0; k < 12; ++k) acc += xr[k] * W[k * DD + d];
    float inv = __frsqrt_rn(v[d] + 1e-5f);
    acc = (acc - m[d]) * inv * g[d] + be[d];
    acc = fmaxf(acc, 0.0f);
    outb[idx] = (__bf16)acc;
}

// ---------------- 128-wide WMMA GEMM, out = act(bn(in @ W + bias))
// Role-swapped tiles: A = W^T (LDS, bank-pad), B = activation rows (global v16bf).
// C/D: lane = output row, VGPR index = output column -> packed row-contiguous stores.
template <bool DO_BN, bool DO_RELU, bool OUTB, bool BOUNDS>
__global__ void gemm128_kernel(const __bf16* __restrict__ in, const float* __restrict__ W,
                               const float* __restrict__ bias,
                               const float* __restrict__ g, const float* __restrict__ be,
                               const float* __restrict__ m, const float* __restrict__ v,
                               __bf16* __restrict__ outb, float* __restrict__ outf,
                               int nrows, int blockOff) {
    __shared__ __bf16 ldsW[DD * LDW];   // ~34 KB, transposed: ldsW[n*LDW + k] = W[k][n]
    __shared__ float scaleL[DD];
    __shared__ float shiftL[DD];
    int tid = threadIdx.x;
    for (int i = tid; i < DD * DD; i += 256) {
        int k = i >> 7, n = i & 127;
        ldsW[n * LDW + k] = (__bf16)W[i];
    }
    if (tid < DD) {
        float bb = bias[tid];
        if (DO_BN) {
            float inv = __frsqrt_rn(v[tid] + 1e-5f);
            float sc = g[tid] * inv;
            scaleL[tid] = sc;
            shiftL[tid] = be[tid] - m[tid] * sc + bb * sc;
        } else {
            scaleL[tid] = 1.0f;
            shiftL[tid] = bb;
        }
    }

    int wave = tid >> 5;
    int lane = tid & 31;
    int nr = lane & 15;        // N index: output row within tile (also M index for A frags)
    int khalf = lane >> 4;     // K-half selector
    int rowBase = (blockIdx.x + blockOff) * 128 + wave * 16;
    int row = rowBase + nr;
    bool rowOk = !BOUNDS || (row < nrows);

    // B fragments (activations): lane holds row `row`, K = kt*32 + khalf*16 .. +16 (contig 32B)
    v16bf bfr[4];
    v16bf z16 = {};
    if (rowOk) {
        const __bf16* bp = in + (size_t)row * DD + khalf * 16;
#pragma unroll
        for (int kt = 0; kt < 4; ++kt) bfr[kt] = *(const v16bf*)(bp + kt * 32);
    } else {
#pragma unroll
        for (int kt = 0; kt < 4; ++kt) bfr[kt] = z16;
    }

    __syncthreads();

    union Frag { v16bf vv; v8bf h[2]; };
#pragma unroll
    for (int mt = 0; mt < 8; ++mt) {
        v8f c = {};
        // A fragment: rows (cols of W) mt*16 + nr, K = kt*32 + khalf*16 + {0..15}
        const __bf16* ap = ldsW + (mt * 16 + nr) * LDW;
#pragma unroll
        for (int kt = 0; kt < 4; ++kt) {
            Frag af;
            af.h[0] = *(const v8bf*)(ap + kt * 32 + khalf * 16);
            af.h[1] = *(const v8bf*)(ap + kt * 32 + khalf * 16 + 8);
            c = __builtin_amdgcn_wmma_f32_16x16x32_bf16(false, af.vv, false, bfr[kt],
                                                        (short)0, c, false, false);
        }
        if (BOUNDS && !rowOk) continue;   // WMMA above runs with full EXEC; only I/O is masked
        int colBase = mt * 16 + khalf * 8;   // lane's 8 consecutive output columns
        float4 s0 = *(const float4*)&scaleL[colBase];
        float4 s1 = *(const float4*)&scaleL[colBase + 4];
        float4 t0 = *(const float4*)&shiftL[colBase];
        float4 t1 = *(const float4*)&shiftL[colBase + 4];
        float val[8];
        val[0] = fmaf(c[0], s0.x, t0.x);
        val[1] = fmaf(c[1], s0.y, t0.y);
        val[2] = fmaf(c[2], s0.z, t0.z);
        val[3] = fmaf(c[3], s0.w, t0.w);
        val[4] = fmaf(c[4], s1.x, t1.x);
        val[5] = fmaf(c[5], s1.y, t1.y);
        val[6] = fmaf(c[6], s1.z, t1.z);
        val[7] = fmaf(c[7], s1.w, t1.w);
        if (DO_RELU) {
#pragma unroll
            for (int i = 0; i < 8; ++i) val[i] = fmaxf(val[i], 0.0f);
        }
        if (OUTB) {
            v8bf ov;
#pragma unroll
            for (int i = 0; i < 8; ++i) ov[i] = (__bf16)val[i];
            *(v8bf*)(outb + (size_t)row * DD + colBase) = ov;     // one 16B store
        } else {
            float4 o0 = {val[0], val[1], val[2], val[3]};
            float4 o1 = {val[4], val[5], val[6], val[7]};
            float* op = outf + (size_t)row * DD + colBase;
            *(float4*)op = o0;                                     // two 16B stores
            *(float4*)(op + 4) = o1;
        }
    }
}

// host-side dispatch over flag combinations + full/tail split
static void launch_gemm(bool do_bn, bool do_relu, bool out_is_bf16,
                        const __bf16* in, const float* W, const float* bias,
                        const float* g, const float* be, const float* m, const float* v,
                        __bf16* outb, float* outf, hipStream_t stream) {
    const int gFull = NN / 128;            // 1562 full blocks (199936 rows)
    const int tail  = NN - gFull * 128;    // 64 rows -> 1 bounded block

#define GEMM_CASE(BN, RL, OB)                                                                   \
    do {                                                                                        \
        gemm128_kernel<BN, RL, OB, false><<<gFull, 256, 0, stream>>>(                           \
            in, W, bias, g, be, m, v, outb, outf, NN, 0);                                       \
        if (tail > 0)                                                                           \
            gemm128_kernel<BN, RL, OB, true><<<1, 256, 0, stream>>>(                            \
                in, W, bias, g, be, m, v, outb, outf, NN, gFull);                               \
    } while (0)

    if (out_is_bf16) {
        if (do_bn) { if (do_relu) GEMM_CASE(true, true, true);  else GEMM_CASE(true, false, true); }
        else       { if (do_relu) GEMM_CASE(false, true, true); else GEMM_CASE(false, false, true); }
    } else {
        if (do_bn) { if (do_relu) GEMM_CASE(true, true, false);  else GEMM_CASE(true, false, false); }
        else       { if (do_relu) GEMM_CASE(false, true, false); else GEMM_CASE(false, false, false); }
    }
#undef GEMM_CASE
}

// ---------------- zero the aggregation buffer
__global__ void zero_kernel(float* __restrict__ p, int n) {
    int i = blockIdx.x * 256 + threadIdx.x;
    if (i < n) p[i] = 0.0f;
}

// ---------------- per-edge: msg = relu(h[src] + bond_lut) ; atomicAdd into agg[dst]
__global__ void edge_kernel(const int* __restrict__ ei, const int* __restrict__ ea,
                            const float* __restrict__ bond,   // [3, 8, 128] for this layer
                            const float* __restrict__ hf, float* __restrict__ agg, int ne) {
    __shared__ float lut[3 * 8 * DD];   // 12 KB
    for (int i = threadIdx.x; i < 3 * 8 * DD; i += 256) lut[i] = bond[i];
    __syncthreads();
    int t = blockIdx.x * 256 + threadIdx.x;
    if (t >= ne * 32) return;
    int e = t >> 5;
    int c = (t & 31) << 2;               // 4-float feature chunk
    int s = ei[e];
    int dsti = ei[ne + e];
    int a0 = ea[e * 3 + 0], a1 = ea[e * 3 + 1], a2 = ea[e * 3 + 2];
    float4 hv = *(const float4*)(hf + (size_t)s * DD + c);
    const float* l0 = lut + a0 * DD + c;
    const float* l1 = lut + 1024 + a1 * DD + c;
    const float* l2 = lut + 2048 + a2 * DD + c;
    float4 rmsg;
    rmsg.x = fmaxf(hv.x + l0[0] + l1[0] + l2[0], 0.0f);
    rmsg.y = fmaxf(hv.y + l0[1] + l1[1] + l2[1], 0.0f);
    rmsg.z = fmaxf(hv.z + l0[2] + l1[2] + l2[2], 0.0f);
    rmsg.w = fmaxf(hv.w + l0[3] + l1[3] + l2[3], 0.0f);
    float* ag = agg + (size_t)dsti * DD + c;
    atomicAdd(ag + 0, rmsg.x);
    atomicAdd(ag + 1, rmsg.y);
    atomicAdd(ag + 2, rmsg.z);
    atomicAdd(ag + 3, rmsg.w);
}

// ---------------- z = (1+eps[l])*h + agg -> bf16
__global__ void combine_kernel(const float* __restrict__ hf, const float* __restrict__ agg,
                               const float* __restrict__ epsp, int l,
                               __bf16* __restrict__ zb, int n) {
    int i = blockIdx.x * 256 + threadIdx.x;
    if (i >= n) return;
    float ep = 1.0f + epsp[l];
    zb[i] = (__bf16)(ep * hf[i] + agg[i]);
}

extern "C" void kernel_launch(void* const* d_in, const int* in_sizes, int n_in,
                              void* d_out, int out_size, void* d_ws, size_t ws_size,
                              hipStream_t stream) {
    (void)in_sizes; (void)n_in; (void)out_size; (void)ws_size;
    const float* x        = (const float*)d_in[0];
    const int*   ei       = (const int*)d_in[1];
    const int*   ea       = (const int*)d_in[2];
    const float* ae_W1    = (const float*)d_in[3];
    const float* ae_b1    = (const float*)d_in[4];
    const float* ae_g1    = (const float*)d_in[5];
    const float* ae_be1   = (const float*)d_in[6];
    const float* ae_m1    = (const float*)d_in[7];
    const float* ae_v1    = (const float*)d_in[8];
    const float* ae_W2    = (const float*)d_in[9];
    const float* ae_b2    = (const float*)d_in[10];
    const float* ae_g2    = (const float*)d_in[11];
    const float* ae_be2   = (const float*)d_in[12];
    const float* ae_m2    = (const float*)d_in[13];
    const float* ae_v2    = (const float*)d_in[14];
    const float* ae_W3    = (const float*)d_in[15];
    const float* ae_b3    = (const float*)d_in[16];
    const float* bond_emb = (const float*)d_in[17];   // [5,3,8,128]
    const float* epsp     = (const float*)d_in[18];   // [5]
    const float* mlp_W1   = (const float*)d_in[19];   // [5,128,128]
    const float* mlp_b1   = (const float*)d_in[20];   // [5,128]
    const float* mlp_g    = (const float*)d_in[21];
    const float* mlp_be   = (const float*)d_in[22];
    const float* mlp_m    = (const float*)d_in[23];
    const float* mlp_v    = (const float*)d_in[24];
    const float* mlp_W2   = (const float*)d_in[25];
    const float* mlp_b2   = (const float*)d_in[26];
    const float* out_g    = (const float*)d_in[27];
    const float* out_be   = (const float*)d_in[28];
    const float* out_m    = (const float*)d_in[29];
    const float* out_v    = (const float*)d_in[30];

    // workspace layout: fp32 h | fp32 agg | bf16 ping | bf16 pong  (~307 MB)
    float*  hf  = (float*)d_ws;
    float*  agg = hf + (size_t)NN * DD;
    __bf16* b0  = (__bf16*)(agg + (size_t)NN * DD);
    __bf16* b1  = b0 + (size_t)NN * DD;

    const int elemN = NN * DD;
    const int gElem = (elemN + 255) / 256;
    const int gEdge = (EE * 32 + 255) / 256;

    // ---- atom encoder
    atom1_kernel<<<gElem, 256, 0, stream>>>(x, ae_W1, ae_b1, ae_g1, ae_be1, ae_m1, ae_v1, b0, NN);
    launch_gemm(true, true, true, b0, ae_W2, ae_b2, ae_g2, ae_be2, ae_m2, ae_v2,
                b1, nullptr, stream);
    launch_gemm(false, false, false, b1, ae_W3, ae_b3, nullptr, nullptr, nullptr, nullptr,
                nullptr, hf, stream);

    // ---- GIN layers
    for (int l = 0; l < LL; ++l) {
        zero_kernel<<<gElem, 256, 0, stream>>>(agg, elemN);
        edge_kernel<<<gEdge, 256, 0, stream>>>(ei, ea, bond_emb + (size_t)l * 3 * 8 * DD,
                                               hf, agg, EE);
        combine_kernel<<<gElem, 256, 0, stream>>>(hf, agg, epsp, l, b0, elemN);
        launch_gemm(true, true, true, b0, mlp_W1 + (size_t)l * DD * DD,
                    mlp_b1 + (size_t)l * DD,
                    mlp_g + (size_t)l * DD, mlp_be + (size_t)l * DD,
                    mlp_m + (size_t)l * DD, mlp_v + (size_t)l * DD,
                    b1, nullptr, stream);
        float* fout = (l == LL - 1) ? (float*)d_out : hf;
        launch_gemm(true, (l < LL - 1), false, b1, mlp_W2 + (size_t)l * DD * DD,
                    mlp_b2 + (size_t)l * DD,
                    out_g + (size_t)l * DD, out_be + (size_t)l * DD,
                    out_m + (size_t)l * DD, out_v + (size_t)l * DD,
                    nullptr, fout, stream);
    }
}
